// Model_65996467471172
// MI455X (gfx1250) — matrix operand order repaired and checked by reading, not compiled, not run on hardware
//
#include <hip/hip_runtime.h>
#include <hip/hip_bf16.h>

// ---------------- problem constants ----------------
constexpr int Cc   = 8;     // clusters
constexpr int Ss   = 500;   // tokens
constexpr int SP   = 512;   // tokens padded
constexpr int DLLM = 768;
constexpr int Bb   = 128;
constexpr int Pp   = 64;
constexpr int DMOD = 512;
constexpr int Hh   = 8;
constexpr int Ee   = 64;
constexpr int HKk  = 512;   // H*E

typedef _Float16 v8h  __attribute__((ext_vector_type(8)));
typedef _Float16 v16h __attribute__((ext_vector_type(16)));
typedef float    v8f  __attribute__((ext_vector_type(8)));

// ---------------- WMMA helpers ----------------
// A fragment (16x32 f16): lane L holds row m=L%16; half=L/16.
// K halfs: [k0+8*half .. +7] and [k0+16+8*half .. +7]
__device__ __forceinline__ v16h ld_fragA(const _Float16* p) {
    const v8h* q = (const v8h*)p;
    v8h lo = q[0];
    v8h hi = q[2];   // +16 halfs
    return __builtin_shufflevector(lo, hi, 0, 1, 2, 3, 4, 5, 6, 7,
                                   8, 9, 10, 11, 12, 13, 14, 15);
}
// B fragment (32x16 f16) from W^T rows: lane L holds col n=L%16;
// K halfs: [k0+8*half .. +7] and [k0+16+8*half .. +7]
__device__ __forceinline__ v16h ld_fragB(const _Float16* p) {
    const v8h* q = (const v8h*)p;
    v8h lo = q[0];
    v8h hi = q[2];   // +16 halfs
    return __builtin_shufflevector(lo, hi, 0, 1, 2, 3, 4, 5, 6, 7,
                                   8, 9, 10, 11, 12, 13, 14, 15);
}
__device__ __forceinline__ v8f wmma16(v16h a, v16h b, v8f c) {
    return __builtin_amdgcn_wmma_f32_16x16x32_f16(false, a, false, b, (short)0, c,
                                                  false, false);
}
__device__ __forceinline__ v8f vzero8() {
    v8f z = {0.f, 0.f, 0.f, 0.f, 0.f, 0.f, 0.f, 0.f};
    return z;
}

// ---------------- stage 1: weight transpose + fp32 -> f16 ----------------
__global__ void transpose_w_kernel(const float* __restrict__ in, _Float16* __restrict__ out,
                                   int K, int N) {
    __shared__ float tile[32][33];
    const int c  = blockIdx.z;
    const int n0 = blockIdx.x * 32;
    const int k0 = blockIdx.y * 32;
    const float* src = in + (size_t)c * K * N;
    _Float16* dst = out + (size_t)c * N * K;
#pragma unroll
    for (int r = threadIdx.y; r < 32; r += 8)
        tile[r][threadIdx.x] = src[(size_t)(k0 + r) * N + (n0 + threadIdx.x)];
    __syncthreads();
#pragma unroll
    for (int r = threadIdx.y; r < 32; r += 8)
        dst[(size_t)(n0 + r) * K + (k0 + threadIdx.x)] = (_Float16)tile[threadIdx.x][r];
}

// ---------------- stage 2: K/V projections ----------------
// grid (SP/64, C), block 256.  K -> Kf16[c][s][hk] ; V -> Vt[c][hk][s]
constexpr int KV_LDST = DLLM + 8;   // 776 halfs

__global__ void kv_kernel(const float* __restrict__ topk,
                          const _Float16* __restrict__ WkT, const _Float16* __restrict__ WvT,
                          const float* __restrict__ bk, const float* __restrict__ bv,
                          _Float16* __restrict__ Kf16, _Float16* __restrict__ Vt) {
    __shared__ __align__(16) _Float16 Xs[64 * KV_LDST];
    const int c    = blockIdx.y;
    const int s0   = blockIdx.x * 64;
    const int wave = threadIdx.x >> 5;
    const int lane = threadIdx.x & 31;
    const int lr   = lane & 15, lh = lane >> 4;
    const int mt   = wave & 3;
    const int nset = wave >> 2;

    // load full X tile [64][768] f32 -> f16 LDS (rows >= S zero-filled)
    for (int t = threadIdx.x; t < 64 * (DLLM / 4); t += 256) {
        int r  = t / (DLLM / 4);
        int q4 = t % (DLLM / 4);
        int s  = s0 + r;
        float4 v = make_float4(0.f, 0.f, 0.f, 0.f);
        if (s < Ss)
            v = ((const float4*)(topk + ((size_t)c * Ss + s) * DLLM + q4 * 4))[0];
        _Float16* dp = &Xs[r * KV_LDST + q4 * 4];
        dp[0] = (_Float16)v.x; dp[1] = (_Float16)v.y;
        dp[2] = (_Float16)v.z; dp[3] = (_Float16)v.w;
    }
    __syncthreads();

    const _Float16* abase = &Xs[(mt * 16 + lr) * KV_LDST + 8 * lh];

    for (int pass = 0; pass < 2; ++pass) {
        const _Float16* WT = (pass ? WvT : WkT) + (size_t)c * HKk * DLLM;
        const float* bias = (pass ? bv : bk) + (size_t)c * HKk;
        for (int nb = 0; nb < 2; ++nb) {
            const int nbase = (nset * 16 + nb * 8) * 16 + lr;  // first n of 8 tiles
            const _Float16* bbase = WT + (size_t)nbase * DLLM + 8 * lh;
            v8f acc[8];
#pragma unroll
            for (int j = 0; j < 8; ++j) acc[j] = vzero8();
            for (int ks = 0; ks < DLLM; ks += 32) {
                v16h a = ld_fragA(abase + ks);
#pragma unroll
                for (int j = 0; j < 8; ++j) {
                    v16h bf = ld_fragB(bbase + j * (16 * DLLM) + ks);
                    acc[j] = wmma16(a, bf, acc[j]);
                }
            }
#pragma unroll
            for (int j = 0; j < 8; ++j) {
                int n = nbase + j * 16;
                float bb = bias[n];
                if (pass == 0) {
#pragma unroll
                    for (int i = 0; i < 8; ++i) {
                        int s = s0 + mt * 16 + 8 * lh + i;
                        Kf16[((size_t)c * SP + s) * HKk + n] = (_Float16)(acc[j][i] + bb);
                    }
                } else {
#pragma unroll
                    for (int i = 0; i < 8; ++i) {
                        int s = s0 + mt * 16 + 8 * lh + i;
                        Vt[((size_t)c * HKk + n) * SP + s] = (_Float16)(acc[j][i] + bb);
                    }
                }
            }
        }
    }
}

// ---------------- stage 3: Q projection ----------------
// grid (B, C), block 256.  Q -> Qf16[c][b*64+p][hk]
constexpr int Q_LDST = DMOD + 8;    // 520 halfs

__global__ void q_kernel(const float* __restrict__ ts, const _Float16* __restrict__ WqT,
                         const float* __restrict__ bq, _Float16* __restrict__ Qf16) {
    __shared__ __align__(16) _Float16 Xs[64 * Q_LDST];
    const int b    = blockIdx.x;
    const int c    = blockIdx.y;
    const int wave = threadIdx.x >> 5;
    const int lane = threadIdx.x & 31;
    const int lr   = lane & 15, lh = lane >> 4;
    const int mt   = wave & 3;
    const int nset = wave >> 2;
    const _Float16* WT = WqT + (size_t)c * HKk * DMOD;

    for (int t = threadIdx.x; t < 64 * (DMOD / 4); t += 256) {
        int r  = t / (DMOD / 4);
        int q4 = t % (DMOD / 4);
        float4 v = ((const float4*)(ts + ((size_t)b * Pp + r) * DMOD + q4 * 4))[0];
        _Float16* dp = &Xs[r * Q_LDST + q4 * 4];
        dp[0] = (_Float16)v.x; dp[1] = (_Float16)v.y;
        dp[2] = (_Float16)v.z; dp[3] = (_Float16)v.w;
    }
    __syncthreads();

    const _Float16* abase = &Xs[(mt * 16 + lr) * Q_LDST + 8 * lh];
    const float* bias = bq + (size_t)c * HKk;

    for (int nb = 0; nb < 2; ++nb) {
        const int nbase = (nset * 16 + nb * 8) * 16 + lr;
        const _Float16* bbase = WT + (size_t)nbase * DMOD + 8 * lh;
        v8f acc[8];
#pragma unroll
        for (int j = 0; j < 8; ++j) acc[j] = vzero8();
        for (int ks = 0; ks < DMOD; ks += 32) {
            v16h a = ld_fragA(abase + ks);
#pragma unroll
            for (int j = 0; j < 8; ++j) {
                v16h bf = ld_fragB(bbase + j * (16 * DMOD) + ks);
                acc[j] = wmma16(a, bf, acc[j]);
            }
        }
#pragma unroll
        for (int j = 0; j < 8; ++j) {
            int n = nbase + j * 16;
            float bb = bias[n];
#pragma unroll
            for (int i = 0; i < 8; ++i) {
                int p = mt * 16 + 8 * lh + i;
                Qf16[((size_t)c * Bb * Pp + b * 64 + p) * HKk + n] = (_Float16)(acc[j][i] + bb);
            }
        }
    }
}

// ---------------- stage 4: attention + gating ----------------
// grid (B, C), block 256.  G[c][b*64+p][hk] = probs[b,p,c] * softmax(QK^T/8) V
constexpr int ATT_LDST = 264;

__global__ void attn_kernel(const _Float16* __restrict__ Q, const _Float16* __restrict__ K,
                            const _Float16* __restrict__ Vt, const float* __restrict__ probs,
                            _Float16* __restrict__ G) {
    __shared__ __align__(16) _Float16 attnS[2 * 64 * ATT_LDST];
    __shared__ float redmax[64];
    __shared__ float redsum[64];

    const int b    = blockIdx.x;
    const int c    = blockIdx.y;
    const int wave = threadIdx.x >> 5;
    const int lane = threadIdx.x & 31;
    const int lr   = lane & 15, lh = lane >> 4;
    const int mt   = wave & 3;       // query m-tile
    const int sh   = wave >> 2;      // s-half
    const float scl = 0.125f;

    const _Float16* Qc = Q + ((size_t)c * Bb * Pp + b * 64) * HKk;
    const _Float16* Kc = K + (size_t)c * SP * HKk;
    const _Float16* Vc = Vt + (size_t)c * HKk * SP;
    const _Float16* qrow = Qc + (size_t)(mt * 16 + lr) * HKk + 8 * lh;

    // softmax sweep assignment: thread owns row (t>>2), col quarter (t&3)*128
    const int srow = threadIdx.x >> 2;
    const int sqc  = threadIdx.x & 3;
    _Float16* sseg = &attnS[(((sqc >> 1) * 64) + srow) * ATT_LDST + (sqc & 1) * 128];

    for (int h = 0; h < Hh; ++h) {
        // ---- scores in two 8-tile chunks; park scaled/masked f16 in LDS ----
        for (int nb = 0; nb < 2; ++nb) {
            const int stile0 = sh * 16 + nb * 8;
            const _Float16* bbase = Kc + (size_t)(stile0 * 16 + lr) * HKk + h * 64 + 8 * lh;
            v8f sc[8];
#pragma unroll
            for (int j = 0; j < 8; ++j) sc[j] = vzero8();
#pragma unroll
            for (int ke = 0; ke < Ee; ke += 32) {
                v16h a = ld_fragA(qrow + h * 64 + ke);
#pragma unroll
                for (int j = 0; j < 8; ++j) {
                    v16h bf = ld_fragB(bbase + j * (16 * HKk) + ke);
                    sc[j] = wmma16(a, bf, sc[j]);
                }
            }
#pragma unroll
            for (int j = 0; j < 8; ++j) {
                int scol = (stile0 + j) * 16 + lr;       // global s index
                bool oob = (scol >= Ss);
                int lcol = scol & 255;
#pragma unroll
                for (int i = 0; i < 8; ++i) {
                    int row = mt * 16 + 8 * lh + i;
                    float v = oob ? -60000.f : sc[j][i] * scl;
                    attnS[(sh * 64 + row) * ATT_LDST + lcol] = (_Float16)v;
                }
            }
        }
        __syncthreads();
        // ---- row max over 512 cols (thread owns 128-col segment) ----
        {
            float m = -3.0e38f;
#pragma unroll
            for (int k = 0; k < 128; ++k) m = fmaxf(m, (float)sseg[k]);
            m = fmaxf(m, __shfl_xor(m, 1, 32));
            m = fmaxf(m, __shfl_xor(m, 2, 32));
            if (sqc == 0) redmax[srow] = m;
        }
        __syncthreads();
        // ---- exp + row sum; overwrite LDS with un-normalized exp ----
        {
            float rm = redmax[srow];
            float ssum = 0.f;
#pragma unroll
            for (int k = 0; k < 128; ++k) {
                float e = __expf((float)sseg[k] - rm);
                ssum += e;
                sseg[k] = (_Float16)e;
            }
            ssum += __shfl_xor(ssum, 1, 32);
            ssum += __shfl_xor(ssum, 2, 32);
            if (sqc == 0) redsum[srow] = ssum;
        }
        __syncthreads();
        // ---- PV: wave owns m-tile mt, two e-tiles ----
        v8f ov[2];
        ov[0] = vzero8(); ov[1] = vzero8();
        const _Float16* vbase = Vc + (size_t)(h * 64 + (sh * 2) * 16 + lr) * SP + 8 * lh;
#pragma unroll
        for (int half = 0; half < 2; ++half) {
            const _Float16* arow = &attnS[(half * 64 + mt * 16 + lr) * ATT_LDST + 8 * lh];
            for (int ks = 0; ks < 256; ks += 32) {
                v16h a = ld_fragA(arow + ks);
#pragma unroll
                for (int jj = 0; jj < 2; ++jj) {
                    v16h bf = ld_fragB(vbase + jj * (16 * SP) + half * 256 + ks);
                    ov[jj] = wmma16(a, bf, ov[jj]);
                }
            }
        }
        // ---- gate by probs / rowsum, store G ----
        float g[8];
#pragma unroll
        for (int i = 0; i < 8; ++i) {
            int row = mt * 16 + 8 * lh + i;
            float pr = probs[((size_t)b * Pp + row) * Cc + c];
            g[i] = pr / redsum[row];
        }
#pragma unroll
        for (int jj = 0; jj < 2; ++jj) {
            int e = (sh * 2 + jj) * 16 + lr;
#pragma unroll
            for (int i = 0; i < 8; ++i) {
                int row = mt * 16 + 8 * lh + i;
                G[((size_t)c * Bb * Pp + b * 64 + row) * HKk + h * 64 + e] =
                    (_Float16)(ov[jj][i] * g[i]);
            }
        }
        __syncthreads();
    }
}

// ---------------- stage 5: out-projection + cluster sum ----------------
// grid (B, 4), block 256; each block does 192 output cols.
__global__ void out_kernel(const _Float16* __restrict__ G, const _Float16* __restrict__ WoT,
                           const float* __restrict__ bo, const float* __restrict__ probs,
                           float* __restrict__ out) {
    const int b    = blockIdx.x;
    const int nq   = blockIdx.y;           // column quarter: 192 cols
    const int wave = threadIdx.x >> 5;
    const int lane = threadIdx.x & 31;
    const int lr   = lane & 15, lh = lane >> 4;
    const int mt   = wave & 3;
    const int nset = wave >> 2;            // 6 n-tiles each
    const int nbase = nq * 192 + nset * 96 + lr;

    v8f acc[6];
#pragma unroll
    for (int j = 0; j < 6; ++j) acc[j] = vzero8();

    for (int c = 0; c < Cc; ++c) {
        const _Float16* arow =
            G + ((size_t)c * Bb * Pp + b * 64 + mt * 16 + lr) * HKk + 8 * lh;
        const _Float16* bbase =
            WoT + (size_t)c * DLLM * HKk + (size_t)nbase * HKk + 8 * lh;
        for (int ks = 0; ks < HKk; ks += 32) {
            v16h a = ld_fragA(arow + ks);
#pragma unroll
            for (int j = 0; j < 6; ++j) {
                v16h bf = ld_fragB(bbase + j * (16 * HKk) + ks);
                acc[j] = wmma16(a, bf, acc[j]);
            }
        }
    }
#pragma unroll
    for (int j = 0; j < 6; ++j) {
        int n = nbase + j * 16;
        float boc[8];
#pragma unroll
        for (int c = 0; c < 8; ++c) boc[c] = bo[(size_t)c * DLLM + n];
#pragma unroll
        for (int i = 0; i < 8; ++i) {
            int row = mt * 16 + 8 * lh + i;
            const float* pr = probs + ((size_t)b * Pp + row) * Cc;
            float4 p0 = ((const float4*)pr)[0];
            float4 p1 = ((const float4*)pr)[1];
            float bias = p0.x * boc[0] + p0.y * boc[1] + p0.z * boc[2] + p0.w * boc[3] +
                         p1.x * boc[4] + p1.y * boc[5] + p1.z * boc[6] + p1.w * boc[7];
            out[((size_t)b * Pp + row) * DLLM + n] = acc[j][i] + bias;
        }
    }
}

// ---------------- host launch ----------------
extern "C" void kernel_launch(void* const* d_in, const int* in_sizes, int n_in,
                              void* d_out, int out_size, void* d_ws, size_t ws_size,
                              hipStream_t stream) {
    (void)in_sizes; (void)n_in; (void)out_size; (void)ws_size;
    const float* topk  = (const float*)d_in[0];
    const float* ts    = (const float*)d_in[1];
    const float* probs = (const float*)d_in[2];
    const float* Wq    = (const float*)d_in[3];
    const float* bq    = (const float*)d_in[4];
    const float* Wk    = (const float*)d_in[5];
    const float* bk    = (const float*)d_in[6];
    const float* Wv    = (const float*)d_in[7];
    const float* bv    = (const float*)d_in[8];
    const float* Wo    = (const float*)d_in[9];
    const float* bo    = (const float*)d_in[10];
    float* out = (float*)d_out;

    _Float16* ws = (_Float16*)d_ws;
    const size_t nWqT = (size_t)Cc * HKk * DMOD;
    const size_t nWkT = (size_t)Cc * HKk * DLLM;
    const size_t nWoT = (size_t)Cc * DLLM * HKk;
    const size_t nKV  = (size_t)Cc * SP * HKk;
    const size_t nQ   = (size_t)Cc * Bb * Pp * HKk;
    _Float16* WqT = ws;
    _Float16* WkT = WqT + nWqT;
    _Float16* WvT = WkT + nWkT;
    _Float16* WoT = WvT + nWkT;
    _Float16* Kf  = WoT + nWoT;
    _Float16* Vt  = Kf + nKV;
    _Float16* Qf  = Vt + nKV;
    _Float16* Gf  = Qf + nQ;

    dim3 tb(32, 8);
    transpose_w_kernel<<<dim3(HKk / 32, DMOD / 32, Cc), tb, 0, stream>>>(Wq, WqT, DMOD, HKk);
    transpose_w_kernel<<<dim3(HKk / 32, DLLM / 32, Cc), tb, 0, stream>>>(Wk, WkT, DLLM, HKk);
    transpose_w_kernel<<<dim3(HKk / 32, DLLM / 32, Cc), tb, 0, stream>>>(Wv, WvT, DLLM, HKk);
    transpose_w_kernel<<<dim3(DLLM / 32, HKk / 32, Cc), tb, 0, stream>>>(Wo, WoT, HKk, DLLM);

    kv_kernel<<<dim3(SP / 64, Cc), 256, 0, stream>>>(topk, WkT, WvT, bk, bv, Kf, Vt);
    q_kernel<<<dim3(Bb, Cc), 256, 0, stream>>>(ts, WqT, bq, Qf);
    attn_kernel<<<dim3(Bb, Cc), 256, 0, stream>>>(Qf, Kf, Vt, probs, Gf);
    out_kernel<<<dim3(Bb, 4), 256, 0, stream>>>(Gf, WoT, bo, probs, out);
}
